// Attention_43276090475231
// MI455X (gfx1250) — matrix operand order repaired and checked by reading, not compiled, not run on hardware
//
#include <hip/hip_runtime.h>
#include <hip/hip_bf16.h>

typedef __bf16 bf16;
typedef __attribute__((ext_vector_type(16))) __bf16 v16bf;
typedef __attribute__((ext_vector_type(8)))  float  v8f;

union Frag { v16bf v; uint4 q[2]; };

// Problem constants
constexpr int B_  = 4;
constexpr int SQ_ = 1024;
constexpr int T_  = B_ * SQ_;      // 4096
constexpr int D_  = 4096;
constexpr int N_  = 32;
constexpr int K_  = 8;
constexpr int G_  = N_ / K_;       // 4
constexpr int H_  = 128;
constexpr int NH_ = N_ * H_;       // 4096
constexpr int KH_ = K_ * H_;       // 1024

// ---------------------------------------------------------------------------
// Pack / transpose kernels (fp32 -> bf16)
// ---------------------------------------------------------------------------
__global__ __launch_bounds__(256) void cast_bf16_kernel(const float* __restrict__ in,
                                                        bf16* __restrict__ out, int n) {
  int idx = blockIdx.x * 256 + threadIdx.x;
  if (idx < n) out[idx] = (bf16)in[idx];
}

// in: [R, C] row-major fp32 -> out: [C, R] row-major bf16.  C = 1<<Cshift.
__global__ __launch_bounds__(256) void transpose_cast_kernel(const float* __restrict__ in,
                                                             bf16* __restrict__ out,
                                                             int R, int Cshift) {
  size_t idx = (size_t)blockIdx.x * 256 + threadIdx.x;
  size_t total = (size_t)R << Cshift;
  if (idx >= total) return;
  int C = 1 << Cshift;
  int r = (int)(idx >> Cshift);
  int c = (int)(idx & (C - 1));
  out[(size_t)c * R + r] = (bf16)in[idx];
}

// ---------------------------------------------------------------------------
// bf16 WMMA GEMM: C[M,Nc] = A[M,Kd] * B[Kd,Nc], Bt stored [Nc,Kd].
// Block = 128 threads (4 waves). Block tile 32(m) x 256(n); each wave owns a
// 32x64 strip (2 M-tiles x 4 N-tiles = 8 wmmas per K=32 step).
// The shared 32x32 A tile is staged in LDS with double-buffered
// global_load_async_to_lds_b128 (ASYNCcnt), one b128 per thread per step.
// ---------------------------------------------------------------------------
__global__ __launch_bounds__(128) void gemm_bf16_kernel(const bf16* __restrict__ A,
                                                        const bf16* __restrict__ Bt,
                                                        float* __restrict__ C,
                                                        int M, int Kd, int Nc) {
  __shared__ __align__(16) bf16 asmem[2][32 * 32];   // double-buffered A tile (2 x 2KB)

  const int tid    = threadIdx.x;
  const int lane   = tid & 31;
  const int wave   = tid >> 5;
  const int m0     = blockIdx.x * 32;
  const int n0     = blockIdx.y * 256 + wave * 64;
  const int half16 = lane >> 4;
  const int l15    = lane & 15;
  const int akb    = half16 * 8;     // A k-base per ISA A-fragment layout

  // async A-tile loader mapping: thread -> (row lm, 8-half chunk lk)
  const int lm = tid >> 2;           // 0..31
  const int lk = (tid & 3) * 8;      // 0,8,16,24 halves
  const bf16* __restrict__ ag = A + (size_t)(m0 + lm) * Kd + lk;

  const bf16* __restrict__ Bp0 = Bt + (size_t)(n0 +  0 + l15) * Kd + half16 * 8;
  const bf16* __restrict__ Bp1 = Bt + (size_t)(n0 + 16 + l15) * Kd + half16 * 8;
  const bf16* __restrict__ Bp2 = Bt + (size_t)(n0 + 32 + l15) * Kd + half16 * 8;
  const bf16* __restrict__ Bp3 = Bt + (size_t)(n0 + 48 + l15) * Kd + half16 * 8;

  const v8f vzero = {0,0,0,0,0,0,0,0};
  v8f acc[2][4];
#pragma unroll
  for (int mt = 0; mt < 2; ++mt)
#pragma unroll
    for (int nt = 0; nt < 4; ++nt) acc[mt][nt] = vzero;

  // prologue: async-load k-step 0 into buffer 0
  {
    unsigned lds0 = (unsigned)(uintptr_t)&asmem[0][lm * 32 + lk];
    unsigned long long ga = (unsigned long long)(uintptr_t)ag;
    asm volatile("global_load_async_to_lds_b128 %0, %1, off"
                 :: "v"(lds0), "v"(ga) : "memory");
  }

  const int nsteps = Kd >> 5;
  for (int step = 0; step < nsteps; ++step) {
    const int kk  = step << 5;
    const int cur = step & 1;

    if (step + 1 < nsteps) {
      // issue next A tile into the other buffer, then wait for current tile
      unsigned ldsn = (unsigned)(uintptr_t)&asmem[cur ^ 1][lm * 32 + lk];
      unsigned long long ga = (unsigned long long)(uintptr_t)(ag + kk + 32);
      asm volatile("global_load_async_to_lds_b128 %0, %1, off"
                   :: "v"(ldsn), "v"(ga) : "memory");
      asm volatile("s_wait_asynccnt 1" ::: "memory");
    } else {
      asm volatile("s_wait_asynccnt 0" ::: "memory");
    }
    __syncthreads();                 // current A tile visible to all waves

    if (step + 2 < nsteps) {         // prefetch B two steps ahead
      __builtin_prefetch(Bp0 + kk + 64, 0, 3);
      __builtin_prefetch(Bp1 + kk + 64, 0, 3);
      __builtin_prefetch(Bp2 + kk + 64, 0, 3);
      __builtin_prefetch(Bp3 + kk + 64, 0, 3);
    }

    // A fragments for the two M-tiles from LDS
    Frag a0, a1;
    const bf16* ar0 = &asmem[cur][l15 * 32 + akb];
    const bf16* ar1 = &asmem[cur][(16 + l15) * 32 + akb];
    a0.q[0] = *(const uint4*)(ar0);
    a0.q[1] = *(const uint4*)(ar0 + 16);
    a1.q[0] = *(const uint4*)(ar1);
    a1.q[1] = *(const uint4*)(ar1 + 16);

    Frag b0, b1, b2, b3;
    b0.q[0] = *(const uint4*)(Bp0 + kk); b0.q[1] = *(const uint4*)(Bp0 + kk + 16);
    b1.q[0] = *(const uint4*)(Bp1 + kk); b1.q[1] = *(const uint4*)(Bp1 + kk + 16);
    b2.q[0] = *(const uint4*)(Bp2 + kk); b2.q[1] = *(const uint4*)(Bp2 + kk + 16);
    b3.q[0] = *(const uint4*)(Bp3 + kk); b3.q[1] = *(const uint4*)(Bp3 + kk + 16);

    acc[0][0] = __builtin_amdgcn_wmma_f32_16x16x32_bf16(false, a0.v, false, b0.v, (short)0, acc[0][0], false, false);
    acc[1][0] = __builtin_amdgcn_wmma_f32_16x16x32_bf16(false, a1.v, false, b0.v, (short)0, acc[1][0], false, false);
    acc[0][1] = __builtin_amdgcn_wmma_f32_16x16x32_bf16(false, a0.v, false, b1.v, (short)0, acc[0][1], false, false);
    acc[1][1] = __builtin_amdgcn_wmma_f32_16x16x32_bf16(false, a1.v, false, b1.v, (short)0, acc[1][1], false, false);
    acc[0][2] = __builtin_amdgcn_wmma_f32_16x16x32_bf16(false, a0.v, false, b2.v, (short)0, acc[0][2], false, false);
    acc[1][2] = __builtin_amdgcn_wmma_f32_16x16x32_bf16(false, a1.v, false, b2.v, (short)0, acc[1][2], false, false);
    acc[0][3] = __builtin_amdgcn_wmma_f32_16x16x32_bf16(false, a0.v, false, b3.v, (short)0, acc[0][3], false, false);
    acc[1][3] = __builtin_amdgcn_wmma_f32_16x16x32_bf16(false, a1.v, false, b3.v, (short)0, acc[1][3], false, false);

    __syncthreads();                 // all waves done reading before overwrite
  }

#pragma unroll
  for (int mt = 0; mt < 2; ++mt) {
#pragma unroll
    for (int r = 0; r < 8; ++r) {
      int m = m0 + mt * 16 + r + half16 * 8;
      float* Crow = C + (size_t)m * Nc + n0 + l15;
      Crow[0]  = acc[mt][0][r];
      Crow[16] = acc[mt][1][r];
      Crow[32] = acc[mt][2][r];
      Crow[48] = acc[mt][3][r];
    }
  }
}

// ---------------------------------------------------------------------------
// RoPE on Q: q_pre fp32 [T,N,H] -> qb bf16 [T,N,H]
// ---------------------------------------------------------------------------
__global__ __launch_bounds__(256) void rope_q_kernel(const float* __restrict__ qp,
                                                     const int* __restrict__ pos,
                                                     bf16* __restrict__ qb) {
  int idx = blockIdx.x * 256 + threadIdx.x;          // T*N*64
  if (idx >= T_ * N_ * 64) return;
  int i = idx & 63;
  int n = (idx >> 6) & (N_ - 1);
  int t = idx >> 11;
  float p = (float)pos[t];
  float freq = __expf(-(float)(2 * i) * (9.210340371976184f / (float)H_));
  float s, c;
  __sincosf(p * freq, &s, &c);
  size_t base = ((size_t)t * N_ + n) * H_;
  float x1 = qp[base + i], x2 = qp[base + i + 64];
  qb[base + i]      = (bf16)(x1 * c - x2 * s);
  qb[base + i + 64] = (bf16)(x2 * c + x1 * s);
}

// RoPE on K + scatter: k_pre fp32 [T,K,H] -> kc fp32 (cache) + kb bf16 [B,SQ,K,H]
__global__ __launch_bounds__(256) void rope_k_kernel(const float* __restrict__ kp,
                                                     const int* __restrict__ pos,
                                                     const int* __restrict__ slot,
                                                     float* __restrict__ kc,
                                                     bf16* __restrict__ kbb) {
  int idx = blockIdx.x * 256 + threadIdx.x;          // T*K*64
  if (idx >= T_ * K_ * 64) return;
  int i = idx & 63;
  int k = (idx >> 6) & (K_ - 1);
  int t = idx >> 9;
  float p = (float)pos[t];
  float freq = __expf(-(float)(2 * i) * (9.210340371976184f / (float)H_));
  float s, c;
  __sincosf(p * freq, &s, &c);
  size_t base = ((size_t)t * K_ + k) * H_;
  float x1 = kp[base + i], x2 = kp[base + i + 64];
  float r1 = x1 * c - x2 * s;
  float r2 = x2 * c + x1 * s;
  size_t cbase = ((size_t)slot[t] * K_ + k) * H_;
  kc[cbase + i]      = r1;
  kc[cbase + i + 64] = r2;
  kbb[base + i]      = (bf16)r1;
  kbb[base + i + 64] = (bf16)r2;
}

// Scatter V: v_pre fp32 [T,K,H] -> vc fp32 (cache) + vt bf16 [B,K,H,SQ]
__global__ __launch_bounds__(256) void scatter_v_kernel(const float* __restrict__ vp,
                                                        const int* __restrict__ slot,
                                                        float* __restrict__ vc,
                                                        bf16* __restrict__ vtt) {
  int idx = blockIdx.x * 256 + threadIdx.x;          // T*K*H
  if (idx >= T_ * K_ * H_) return;
  int h = idx & (H_ - 1);
  int k = (idx >> 7) & (K_ - 1);
  int t = idx >> 10;
  float val = vp[idx];
  vc[((size_t)slot[t] * K_ + k) * H_ + h] = val;
  int b = t >> 10;             // / SQ_
  int s = t & (SQ_ - 1);
  vtt[(((size_t)b * K_ + k) * H_ + h) * SQ_ + s] = (bf16)val;
}

// ---------------------------------------------------------------------------
// Flash attention. One wave (32 threads) per (b, head n, 16-query tile).
// Scores: S(16q x 16s) = Q(16xH) * K^T via 4 wmmas (K-dim = H = 128).
// PV: O += P(16x32) * V(32x16) per 16-wide h tile (8 tiles) via wmma.
// ---------------------------------------------------------------------------
__global__ __launch_bounds__(32) void attn_kernel(const bf16* __restrict__ qb,
                                                  const bf16* __restrict__ kbb,
                                                  const bf16* __restrict__ vtt,
                                                  bf16* __restrict__ aout) {
  __shared__ __align__(32) bf16 lds[16 * 32];        // P tile, bf16 [16 q][32 s]

  const int lane   = threadIdx.x & 31;
  const int half16 = lane >> 4;
  const int l15    = lane & 15;
  const int akb    = half16 * 8;

  int w  = blockIdx.x;                               // B * N * (SQ/16) waves
  int qt = w & 63;                                   // SQ/16 = 64
  int n  = (w >> 6) & (N_ - 1);
  int b  = w >> 11;
  int q0 = qt * 16;
  int kh = n / G_;

  // Preload Q A-fragments for the 4 h-chunks (K-dim 32 each)
  Frag Aq[4];
  const bf16* __restrict__ qrow = qb + ((size_t)(b * SQ_ + q0 + l15) * N_ + n) * H_;
#pragma unroll
  for (int c = 0; c < 4; ++c) {
    Aq[c].q[0] = *(const uint4*)(qrow + c * 32 + akb);
    Aq[c].q[1] = *(const uint4*)(qrow + c * 32 + akb + 16);
  }

  const v8f vzero = {0,0,0,0,0,0,0,0};
  v8f acc[8];
#pragma unroll
  for (int i = 0; i < 8; ++i) acc[i] = vzero;
  float rm[8], rs[8];
#pragma unroll
  for (int r = 0; r < 8; ++r) { rm[r] = -1e30f; rs[r] = 0.0f; }

  const float scale = 0.08838834764831845f;          // 1/sqrt(128)

  for (int s0 = 0; s0 < q0 + 16; s0 += 32) {
    // ---- score tiles: s0..s0+15 and s0+16..s0+31 ----
    v8f t0 = vzero, t1 = vzero;
    {
      const bf16* __restrict__ krow0 =
          kbb + ((size_t)(b * SQ_ + s0 + l15) * K_ + kh) * H_ + half16 * 8;
      const bf16* __restrict__ krow1 =
          kbb + ((size_t)(b * SQ_ + s0 + 16 + l15) * K_ + kh) * H_ + half16 * 8;
#pragma unroll
      for (int c = 0; c < 4; ++c) {
        Frag bk0, bk1;
        bk0.q[0] = *(const uint4*)(krow0 + c * 32);
        bk0.q[1] = *(const uint4*)(krow0 + c * 32 + 16);
        bk1.q[0] = *(const uint4*)(krow1 + c * 32);
        bk1.q[1] = *(const uint4*)(krow1 + c * 32 + 16);
        t0 = __builtin_amdgcn_wmma_f32_16x16x32_bf16(false, Aq[c].v, false, bk0.v, (short)0, t0, false, false);
        t1 = __builtin_amdgcn_wmma_f32_16x16x32_bf16(false, Aq[c].v, false, bk1.v, (short)0, t1, false, false);
      }
    }

    // ---- online softmax update ----
#pragma unroll
    for (int r = 0; r < 8; ++r) {
      int qpos = q0 + r + half16 * 8;
      int sa0 = s0 + l15;
      int sa1 = s0 + 16 + l15;
      float v0 = t0[r] * scale;
      float v1 = t1[r] * scale;
      if (sa0 > qpos) v0 = -1e30f;
      if (sa1 > qpos) v1 = -1e30f;
      float cm = fmaxf(v0, v1);
#pragma unroll
      for (int off = 1; off < 16; off <<= 1)
        cm = fmaxf(cm, __shfl_xor(cm, off, 16));
      float nm = fmaxf(rm[r], cm);
      float alpha = __expf(rm[r] - nm);
      rm[r] = nm;
      float p0 = __expf(v0 - nm);
      float p1 = __expf(v1 - nm);
      float ssum = p0 + p1;
#pragma unroll
      for (int off = 1; off < 16; off <<= 1)
        ssum += __shfl_xor(ssum, off, 16);
      rs[r] = rs[r] * alpha + ssum;
#pragma unroll
      for (int ht = 0; ht < 8; ++ht) acc[ht][r] *= alpha;
      int m = r + half16 * 8;
      lds[m * 32 + l15]      = (bf16)p0;
      lds[m * 32 + 16 + l15] = (bf16)p1;
    }
    asm volatile("s_wait_dscnt 0" ::: "memory");

    // ---- P as A-fragment from LDS ----
    Frag Ap;
    const bf16* prow = lds + l15 * 32 + akb;
    Ap.q[0] = *(const uint4*)(prow);
    Ap.q[1] = *(const uint4*)(prow + 16);

    // ---- PV: 8 h tiles of 16 ----
#pragma unroll
    for (int ht = 0; ht < 8; ++ht) {
      const bf16* __restrict__ vrow =
          vtt + (((size_t)(b * K_ + kh) * H_) + ht * 16 + l15) * SQ_ + s0 + half16 * 8;
      Frag bv;
      bv.q[0] = *(const uint4*)(vrow);
      bv.q[1] = *(const uint4*)(vrow + 16);
      acc[ht] = __builtin_amdgcn_wmma_f32_16x16x32_bf16(false, Ap.v, false, bv.v, (short)0, acc[ht], false, false);
    }
  }

  // ---- normalize + store bf16 attn output [T, N, H] ----
#pragma unroll
  for (int r = 0; r < 8; ++r) {
    float inv = 1.0f / rs[r];
    int m = q0 + r + half16 * 8;
    bf16* orow = aout + ((size_t)(b * SQ_ + m) * N_ + n) * H_ + l15;
#pragma unroll
    for (int ht = 0; ht < 8; ++ht) orow[ht * 16] = (bf16)(acc[ht][r] * inv);
  }
}

// ---------------------------------------------------------------------------
// Host-side launcher
// ---------------------------------------------------------------------------
extern "C" void kernel_launch(void* const* d_in, const int* in_sizes, int n_in,
                              void* d_out, int out_size, void* d_ws, size_t ws_size,
                              hipStream_t stream) {
  (void)in_sizes; (void)n_in; (void)out_size; (void)ws_size;

  const float* x    = (const float*)d_in[0];
  const float* w_q  = (const float*)d_in[3];   // [D, N, H]
  const float* w_k  = (const float*)d_in[4];   // [D, K, H]
  const float* w_v  = (const float*)d_in[5];   // [D, K, H]
  const float* w_o  = (const float*)d_in[6];   // [N, H, D]
  const int* positions    = (const int*)d_in[8];
  const int* slot_mapping = (const int*)d_in[9];

  float* out = (float*)d_out;
  float* kc = out;                               // [NUM_PAGES*PAGE, K, H] = T*K*H floats
  float* vc = out + (size_t)T_ * K_ * H_;
  float* o  = out + (size_t)2 * T_ * K_ * H_;    // [T, D]

  // workspace layout
  char* ws = (char*)d_ws;
  size_t off = 0;
  auto take = [&](size_t bytes) { char* p = ws + off; off += bytes; return p; };
  bf16* xb   = (bf16*)take((size_t)T_  * D_  * 2);   // x in bf16
  bf16* wqt  = (bf16*)take((size_t)NH_ * D_  * 2);   // [NH, D]
  bf16* wkt  = (bf16*)take((size_t)KH_ * D_  * 2);   // [KH, D]
  bf16* wvt  = (bf16*)take((size_t)KH_ * D_  * 2);   // [KH, D]
  bf16* wot  = (bf16*)take((size_t)D_  * NH_ * 2);   // [D, NH]
  bf16* qbuf = (bf16*)take((size_t)T_  * NH_ * 2);   // roped Q bf16 [T,N,H]
  bf16* kbuf = (bf16*)take((size_t)T_  * KH_ * 2);   // roped K bf16 [B,SQ,K,H]
  bf16* vtb  = (bf16*)take((size_t)T_  * KH_ * 2);   // V bf16 transposed [B,K,H,SQ]
  bf16* aout = (bf16*)take((size_t)T_  * NH_ * 2);   // attention out bf16 [T,N,H]
  float* qpre = (float*)take((size_t)T_ * NH_ * 4);  // pre-rope Q
  float* kpre = (float*)take((size_t)T_ * KH_ * 4);  // pre-rope K
  float* vpre = (float*)take((size_t)T_ * KH_ * 4);  // V fp32

  // 1) pack activations & weights to bf16 (weights transposed for B-fragments)
  cast_bf16_kernel<<<(T_ * D_) / 256, 256, 0, stream>>>(x, xb, T_ * D_);
  transpose_cast_kernel<<<(D_ * NH_) / 256, 256, 0, stream>>>(w_q, wqt, D_, 12);   // [D,NH]->[NH,D]
  transpose_cast_kernel<<<(D_ * KH_) / 256, 256, 0, stream>>>(w_k, wkt, D_, 10);   // [D,KH]->[KH,D]
  transpose_cast_kernel<<<(D_ * KH_) / 256, 256, 0, stream>>>(w_v, wvt, D_, 10);
  transpose_cast_kernel<<<(NH_ * D_) / 256, 256, 0, stream>>>(w_o, wot, NH_, 12);  // [NH,D]->[D,NH]

  // 2) projection GEMMs (bf16 WMMA, fp32 accumulate; async-LDS A staging)
  gemm_bf16_kernel<<<dim3(T_ / 32, NH_ / 256), 128, 0, stream>>>(xb, wqt, qpre, T_, D_, NH_);
  gemm_bf16_kernel<<<dim3(T_ / 32, KH_ / 256), 128, 0, stream>>>(xb, wkt, kpre, T_, D_, KH_);
  gemm_bf16_kernel<<<dim3(T_ / 32, KH_ / 256), 128, 0, stream>>>(xb, wvt, vpre, T_, D_, KH_);

  // 3) RoPE + cache scatter
  rope_q_kernel<<<(T_ * N_ * 64) / 256, 256, 0, stream>>>(qpre, positions, qbuf);
  rope_k_kernel<<<(T_ * K_ * 64) / 256, 256, 0, stream>>>(kpre, positions, slot_mapping, kc, kbuf);
  scatter_v_kernel<<<(T_ * K_ * H_) / 256, 256, 0, stream>>>(vpre, slot_mapping, vc, vtb);

  // 4) flash attention (one wave per (b, head, 16-query tile))
  attn_kernel<<<B_ * N_ * (SQ_ / 16), 32, 0, stream>>>(qbuf, kbuf, vtb, aout);

  // 5) output projection
  gemm_bf16_kernel<<<dim3(T_ / 32, D_ / 256), 128, 0, stream>>>(aout, wot, o, T_, NH_, D_);
}
